// Attention_49048526520431
// MI455X (gfx1250) — compile-verified, matrix operand order checked by reading, not run on hardware
//
#include <hip/hip_runtime.h>

typedef __attribute__((ext_vector_type(16))) __bf16 v16bf;
typedef __attribute__((ext_vector_type(8)))  __bf16 v8bf;
typedef __attribute__((ext_vector_type(8)))  float  v8f;

#define BATCH 16
#define SEQ   2048
#define DIM   256

union Frag { v16bf v; v8bf h[2]; };

static __device__ __forceinline__ __bf16 f2bf(float f) {
    unsigned u = __builtin_bit_cast(unsigned, f);
    unsigned r = (u + 0x7FFFu + ((u >> 16) & 1u)) >> 16;  // RNE
    unsigned short s = (unsigned short)r;
    return __builtin_bit_cast(__bf16, s);
}

static __device__ __forceinline__ v8f v8f_zero() {
    v8f z;
#pragma unroll
    for (int i = 0; i < 8; ++i) z[i] = 0.0f;
    return z;
}

// ---------------------------------------------------------------------------
// Tensor Data Mover staging (gfx1250). Guarded: 5-arg builtin on clang-22
// (ROCm 7.2), 6-arg on clang-23 (amdgpu-toolchain); manual copy fallback.
// ---------------------------------------------------------------------------
#if defined(__has_builtin)
#if __has_builtin(__builtin_amdgcn_tensor_load_to_lds)
#define HAVE_TDM 1
#endif
#endif

#ifdef HAVE_TDM
typedef __attribute__((ext_vector_type(4))) unsigned tdm_v4u;
typedef __attribute__((ext_vector_type(8))) int      tdm_v8i;
typedef __attribute__((ext_vector_type(4))) int      tdm_v4i;

// Load a 64x256 bf16 tile (row stride 256 elems in memory) into LDS at
// lds_off, padding LDS by 4 DWORDs (16B) after every 128 DWORDs (one 512B
// row) -> 264-element padded rows, matching the fragment-load layout.
static __device__ __forceinline__ void tdm_load_k_tile(unsigned lds_off,
                                                       const __bf16* gptr) {
    unsigned long long ga = (unsigned long long)(uintptr_t)gptr;
    tdm_v4u g0;
    g0[0] = 1u;                                            // count=1 (valid D#)
    g0[1] = lds_off;                                       // LDS byte address
    g0[2] = (unsigned)ga;                                  // global addr lo
    g0[3] = (unsigned)((ga >> 32) & 0x01FFFFFFu) | (2u << 30); // hi | type=2
    tdm_v8i g1;
    g1[0] = (int)((1u << 16) |          // data_size = 2 bytes
                  (1u << 20) |          // pad_enable
                  (6u << 22) |          // pad_interval: every 128 DWORDs
                  (3u << 25));          // pad_amount: 4 DWORDs
    g1[1] = (int)(256u << 16);          // tensor_dim0 = 256 (lo16)
    g1[2] = (int)(64u << 16);           // tensor_dim0 hi=0 | tensor_dim1 = 64
    g1[3] = (int)(256u << 16);          // tensor_dim1 hi=0 | tile_dim0 = 256
    g1[4] = (int)64u;                   // tile_dim1 = 64, tile_dim2 = 0
    g1[5] = (int)256u;                  // tensor_dim0_stride = 256 (lo32)
    g1[6] = 0;                          // stride0 hi | stride1 lo
    g1[7] = 0;                          // stride1 hi
    tdm_v4i z4 = {0, 0, 0, 0};
#if __clang_major__ >= 23
    tdm_v8i z8 = {0, 0, 0, 0, 0, 0, 0, 0};
    __builtin_amdgcn_tensor_load_to_lds(g0, g1, z4, z4, z8, 0);
#else
    __builtin_amdgcn_tensor_load_to_lds(g0, g1, z4, z4, 0);
#endif
}
#endif  // HAVE_TDM

// ---------------------------------------------------------------------------
// Kernel 0: convert + transpose weights: Wt[mat][out][in] (bf16) = W[in][out]
// ---------------------------------------------------------------------------
__global__ __launch_bounds__(256) void wconv_kernel(const float* __restrict__ Wq,
                                                    const float* __restrict__ Wk,
                                                    const float* __restrict__ Wv,
                                                    __bf16* __restrict__ wt) {
    int tid = blockIdx.x * 256 + threadIdx.x;             // 0 .. 196607
    const float* W = (tid < 65536) ? Wq : (tid < 131072 ? Wk : Wv);
    int rem = tid & 65535;
    int o = rem >> 8;
    int i = rem & 255;
    wt[tid] = f2bf(W[i * 256 + o]);
}

// ---------------------------------------------------------------------------
// Kernel 1: QKV projection. Block = one 16-row strip of x, 8 waves.
// Q is pre-scaled by 1/sqrt(D) so the attention loop needs no score scaling.
// ---------------------------------------------------------------------------
__global__ __launch_bounds__(256) void qkv_kernel(const float* __restrict__ x,
                                                  const __bf16* __restrict__ wt,
                                                  const float* __restrict__ bq,
                                                  const float* __restrict__ bk,
                                                  const float* __restrict__ bv,
                                                  __bf16* __restrict__ qws,
                                                  __bf16* __restrict__ kws,
                                                  __bf16* __restrict__ vws) {
    __shared__ __align__(16) __bf16 xl[16][264];          // padded rows

    const int strip = blockIdx.x;                         // 0..2047
    const int t = threadIdx.x;

    {   // stage x strip (16x256 f32) -> bf16 LDS
        int r = t >> 4;
        int c0 = (t & 15) * 16;
        const float* xp = x + ((size_t)strip * 16 + r) * DIM + c0;
#pragma unroll
        for (int e = 0; e < 16; ++e) xl[r][c0 + e] = f2bf(xp[e]);
    }
    __syncthreads();

    const int lane = t & 31;
    const int w = t >> 5;
    const int n = lane & 15;
    const int half = lane >> 4;
    const int kb = half * 8;

    // A fragments (16x32 bf16 per K-chunk), shared across this wave's 6 tiles
    Frag a[8];
#pragma unroll
    for (int dc = 0; dc < 8; ++dc) {
        a[dc].h[0] = *(const v8bf*)&xl[n][dc * 32 + kb];
        a[dc].h[1] = *(const v8bf*)&xl[n][dc * 32 + kb + 16];
    }

#pragma unroll
    for (int mat = 0; mat < 3; ++mat) {
        const float* bptr = (mat == 0) ? bq : (mat == 1) ? bk : bv;
        __bf16* op = (mat == 0) ? qws : (mat == 1) ? kws : vws;
#pragma unroll
        for (int i2 = 0; i2 < 2; ++i2) {
            int ct = w * 2 + i2;                          // 0..15

            v8f c = v8f_zero();
            const __bf16* wrow = wt + ((size_t)mat << 16) +
                                 (size_t)(ct * 16 + n) * 256;
#pragma unroll
            for (int dc = 0; dc < 8; ++dc) {
                Frag bf;
                bf.h[0] = *(const v8bf*)(wrow + dc * 32 + kb);
                bf.h[1] = *(const v8bf*)(wrow + dc * 32 + kb + 16);
                c = __builtin_amdgcn_wmma_f32_16x16x32_bf16(
                        false, a[dc].v, false, bf.v, (short)0, c, false, false);
            }

            float bias = bptr[ct * 16 + n];
#pragma unroll
            for (int j = 0; j < 8; ++j) {
                float val = c[j] + bias;
                if (mat == 0) val *= 0.0625f;             // fold 1/sqrt(256) into Q
                int row = strip * 16 + j + half * 8;      // D-layout rows j/j+8
                op[(size_t)row * DIM + ct * 16 + n] = f2bf(val);
            }
        }
    }
}

// ---------------------------------------------------------------------------
// Kernel 2: flash attention + sum over queries. 64-key tiles.
// Block = 4 waves, same batch, adjacent q-tiles. K tile streamed by the TDM
// (wave 0 issues tensor_load_to_lds with LDS padding; s_wait_tensorcnt),
// V tile transposed into LDS by all threads, next tiles prefetched.
// Accumulator rescale is skipped (wave-uniform branch) when no row max moved.
// ---------------------------------------------------------------------------
__global__ __launch_bounds__(128) void attn_kernel(const __bf16* __restrict__ qws,
                                                   const __bf16* __restrict__ kws,
                                                   const __bf16* __restrict__ vws,
                                                   float* __restrict__ out) {
    __shared__ __align__(16) __bf16 kl[64][264];          // K tile (TDM-padded)
    __shared__ __align__(16) __bf16 vl[256][72];          // V tile transposed
    __shared__ __align__(16) __bf16 pl[4][16][72];        // per-wave P buffer

    const int b = blockIdx.x >> 5;                        // 16 batches
    const int qtg = blockIdx.x & 31;                      // 32 q-tile groups
    const int t = threadIdx.x;
    const int lane = t & 31;
    const int w = t >> 5;
    const int qt = qtg * 4 + w;                           // 0..127

    const int n = lane & 15;
    const int half = lane >> 4;
    const int kb = half * 8;

    // Q fragments (already scaled by 1/sqrt(D)) for this wave's 16 rows
    Frag qf[8];
    {
        const __bf16* qrow = qws + ((size_t)b * SEQ + qt * 16 + n) * DIM;
#pragma unroll
        for (int dc = 0; dc < 8; ++dc) {
            qf[dc].h[0] = *(const v8bf*)(qrow + dc * 32 + kb);
            qf[dc].h[1] = *(const v8bf*)(qrow + dc * 32 + kb + 16);
        }
    }

    v8f acc[16];
#pragma unroll
    for (int tt = 0; tt < 16; ++tt) acc[tt] = v8f_zero();
    float mrun[8], lrun[8];
#pragma unroll
    for (int j = 0; j < 8; ++j) { mrun[j] = -INFINITY; lrun[j] = 0.0f; }

    for (int kt = 0; kt < SEQ / 64; ++kt) {
        __syncthreads();                                  // prev tile consumed
        {
            int r = t >> 1;                               // key row 0..63
            int cb = (t & 1) * 128;                       // d column base
            const __bf16* vp = vws + ((size_t)b * SEQ + kt * 64 + r) * DIM + cb;
#ifdef HAVE_TDM
            if (w == 0) {
                tdm_load_k_tile((unsigned)(uintptr_t)&kl[0][0],
                                kws + ((size_t)b * SEQ + kt * 64) * DIM);
            }
#else
            const __bf16* kp = kws + ((size_t)b * SEQ + kt * 64 + r) * DIM + cb;
#pragma unroll
            for (int i2 = 0; i2 < 16; ++i2)
                *(v8bf*)&kl[r][cb + i2 * 8] = *(const v8bf*)(kp + i2 * 8);
#endif
#pragma unroll
            for (int i2 = 0; i2 < 16; ++i2) {             // V transpose stage
                v8bf vv = *(const v8bf*)(vp + i2 * 8);
#pragma unroll
                for (int e = 0; e < 8; ++e) vl[cb + i2 * 8 + e][r] = vv[e];
            }
            if (kt + 1 < SEQ / 64) {                      // prefetch next tiles
                __builtin_prefetch(vp + 64 * DIM, 0, 1);
                __builtin_prefetch(kws + ((size_t)b * SEQ + (kt + 1) * 64 + r) * DIM + cb, 0, 1);
            }
#ifdef HAVE_TDM
            if (w == 0) __builtin_amdgcn_s_wait_tensorcnt(0);
#endif
        }
        __syncthreads();

        // scores S[16 x 64] = Q · K^T  (four 16x16 N-tiles, K-dim = D)
        v8f s[4];
#pragma unroll
        for (int nt = 0; nt < 4; ++nt) s[nt] = v8f_zero();
#pragma unroll
        for (int dc = 0; dc < 8; ++dc) {
#pragma unroll
            for (int nt = 0; nt < 4; ++nt) {
                Frag bf;
                bf.h[0] = *(const v8bf*)&kl[nt * 16 + n][dc * 32 + kb];
                bf.h[1] = *(const v8bf*)&kl[nt * 16 + n][dc * 32 + kb + 16];
                s[nt] = __builtin_amdgcn_wmma_f32_16x16x32_bf16(
                            false, qf[dc].v, false, bf.v, (short)0, s[nt], false, false);
            }
        }

        // online softmax; rows j (lanes 0-15) / j+8 (lanes 16-31)
        float alpha_arr[8];
        int need = 0;
#pragma unroll
        for (int j = 0; j < 8; ++j) {
            float a0 = s[0][j], a1 = s[1][j], a2 = s[2][j], a3 = s[3][j];
            float mx = fmaxf(fmaxf(a0, a1), fmaxf(a2, a3));
#pragma unroll
            for (int off = 8; off >= 1; off >>= 1) mx = fmaxf(mx, __shfl_xor(mx, off, 32));
            float mnew = fmaxf(mrun[j], mx);
            need |= (mnew > mrun[j]) ? 1 : 0;
            float alpha = __expf(mrun[j] - mnew);
            alpha_arr[j] = alpha;
            float p0 = __expf(a0 - mnew);
            float p1 = __expf(a1 - mnew);
            float p2 = __expf(a2 - mnew);
            float p3 = __expf(a3 - mnew);
            float ls = (p0 + p1) + (p2 + p3);
#pragma unroll
            for (int off = 8; off >= 1; off >>= 1) ls += __shfl_xor(ls, off, 32);
            lrun[j] = lrun[j] * alpha + ls;
            mrun[j] = mnew;
            int row = j + half * 8;
            pl[w][row][n]      = f2bf(p0);
            pl[w][row][n + 16] = f2bf(p1);
            pl[w][row][n + 32] = f2bf(p2);
            pl[w][row][n + 48] = f2bf(p3);
        }
        if (__any(need)) {                                // wave-uniform skip
#pragma unroll
            for (int tt = 0; tt < 16; ++tt)
#pragma unroll
                for (int j = 0; j < 8; ++j) acc[tt][j] *= alpha_arr[j];
        }
        asm volatile("s_wait_dscnt 0" ::: "memory");      // wave-local P visibility

        // P[16x64] as two A-fragments, then acc += P · V
        Frag pf[2];
#pragma unroll
        for (int c = 0; c < 2; ++c) {
            pf[c].h[0] = *(const v8bf*)&pl[w][n][c * 32 + kb];
            pf[c].h[1] = *(const v8bf*)&pl[w][n][c * 32 + kb + 16];
        }
#pragma unroll
        for (int tt = 0; tt < 16; ++tt) {
#pragma unroll
            for (int c = 0; c < 2; ++c) {
                Frag vf;
                vf.h[0] = *(const v8bf*)&vl[tt * 16 + n][c * 32 + kb];
                vf.h[1] = *(const v8bf*)&vl[tt * 16 + n][c * 32 + kb + 16];
                acc[tt] = __builtin_amdgcn_wmma_f32_16x16x32_bf16(
                              false, pf[c].v, false, vf.v, (short)0, acc[tt], false, false);
            }
        }
    }

    // normalize and reduce over query rows -> out[b, d] (atomic accumulate)
#pragma unroll
    for (int tt = 0; tt < 16; ++tt) {
        float part = 0.0f;
#pragma unroll
        for (int j = 0; j < 8; ++j) part += acc[tt][j] / lrun[j];
        atomicAdd(out + b * DIM + tt * 16 + n, part);
    }
}

// ---------------------------------------------------------------------------
extern "C" void kernel_launch(void* const* d_in, const int* in_sizes, int n_in,
                              void* d_out, int out_size, void* d_ws, size_t ws_size,
                              hipStream_t stream) {
    const float* x  = (const float*)d_in[0];
    const float* Wq = (const float*)d_in[1];
    const float* bq = (const float*)d_in[2];
    const float* Wk = (const float*)d_in[3];
    const float* bk = (const float*)d_in[4];
    const float* Wv = (const float*)d_in[5];
    const float* bv = (const float*)d_in[6];
    float* out = (float*)d_out;

    char* ws = (char*)d_ws;
    const size_t QKV_BYTES = (size_t)BATCH * SEQ * DIM * sizeof(unsigned short);
    __bf16* wt  = (__bf16*)ws;                                   // 384 KB used
    __bf16* qws = (__bf16*)(ws + (1u << 19));
    __bf16* kws = (__bf16*)(ws + (1u << 19) + QKV_BYTES);
    __bf16* vws = (__bf16*)(ws + (1u << 19) + 2 * QKV_BYTES);

    hipMemsetAsync(d_out, 0, (size_t)out_size * sizeof(float), stream);

    wconv_kernel<<<768, 256, 0, stream>>>(Wq, Wk, Wv, wt);
    qkv_kernel<<<(BATCH * SEQ) / 16, 256, 0, stream>>>(x, wt, bq, bk, bv, qws, kws, vws);
    attn_kernel<<<BATCH * (SEQ / 16 / 4), 128, 0, stream>>>(qws, kws, vws, out);
}
